// MoETransformerBlock_10050223473273
// MI455X (gfx1250) — matrix operand order repaired and checked by reading, not compiled, not run on hardware
//
#include <hip/hip_runtime.h>
#include <hip/hip_bf16.h>

// ---- problem constants (from reference) ----
#define B_SZ   2
#define LIMG   1024
#define LTXT   512
#define S_TOT  1536     // LIMG + LTXT
#define NHEAD  16
#define NKVH   8
#define HDIM   128
#define DIMG   2048
#define DTXT   1024
#define INNER  2048     // NHEAD*HDIM

typedef __attribute__((ext_vector_type(16))) __bf16 v16bf;
typedef __attribute__((ext_vector_type(8)))  float  v8f;

union FragA {
  v16bf bf;
  unsigned int u[8];
  unsigned short h[16];
};

// native hardware convert (clang selects gfx1250 bf16 cvt ops)
__device__ __forceinline__ unsigned short f2bf(float f) {
  union { __bf16 b; unsigned short s; } cv;
  cv.b = (__bf16)f;
  return cv.s;
}
__device__ __forceinline__ float bf2f(unsigned short h) {
  return __uint_as_float(((unsigned int)h) << 16);
}
// branch-free tanh: 1 - 2/(e^(2x)+1); saturates to +/-1 without NaN
__device__ __forceinline__ float tanh_fast(float x) {
  float e = __expf(2.0f * x);
  return 1.0f - 2.0f / (e + 1.0f);
}
__device__ __forceinline__ float gelu_tanh(float x) {
  float x3 = x * x * x;
  return 0.5f * x * (1.0f + tanh_fast(0.7978845608028654f * (x + 0.044715f * x3)));
}

// ======================================================================
// RMSNorm: y_bf16 = (x [+ add]) * rsqrt(mean(sq)+eps) * (1 + scale)
// ======================================================================
__global__ __launch_bounds__(256) void rmsnorm_kernel(
    const float* __restrict__ X, const float* __restrict__ Add,
    const float* __restrict__ scale, unsigned short* __restrict__ Y, int D) {
  __shared__ float red[256];
  long row = blockIdx.x;
  const float* xr = X + row * (long)D;
  const float* ar = Add ? Add + row * (long)D : nullptr;
  float ss = 0.f;
  for (int i = threadIdx.x; i < D; i += 256) {
    float v = xr[i];
    if (ar) v += ar[i];
    ss += v * v;
  }
  red[threadIdx.x] = ss;
  __syncthreads();
  for (int s = 128; s > 0; s >>= 1) {
    if ((int)threadIdx.x < s) red[threadIdx.x] += red[threadIdx.x + s];
    __syncthreads();
  }
  float rs = rsqrtf(red[0] / (float)D + 1e-6f);
  for (int i = threadIdx.x; i < D; i += 256) {
    float v = xr[i];
    if (ar) v += ar[i];
    Y[row * (long)D + i] = f2bf(v * rs * (1.0f + scale[i]));
  }
}

// ======================================================================
// RoPE (half-split); positions = global joint-sequence index.
// ======================================================================
__global__ __launch_bounds__(64) void rope_kernel(float* __restrict__ X,
                                                  int nheads, float out_scale) {
  int i = threadIdx.x;                // 0..63
  long row = blockIdx.x;              // 0 .. B*S_TOT-1
  int head = blockIdx.y;
  int s = (int)(row % S_TOT);
  float inv_freq = __expf(-logf(10000.0f) * ((float)i / 64.0f));
  float th = (float)s * inv_freq;
  float sn, cs;
  __sincosf(th, &sn, &cs);
  float* p = X + (row * (long)nheads + head) * (long)HDIM;
  float x1 = p[i], x2 = p[i + 64];
  p[i]      = (x1 * cs - x2 * sn) * out_scale;
  p[i + 64] = (x2 * cs + x1 * sn) * out_scale;
}

// ======================================================================
// Generic WMMA bf16 GEMM with double-buffered LDS tiles + weight prefetch.
//   A: bf16 (A1==null) or fused gelu(A0)*A1 (both bf16)
//   B: fp32 weights -> bf16 at LDS stage
//   C: fp32 (Cf, optional fused residual Radd) or bf16 (Cb)
//   Row remap for A and C: row = off + (m/seg)*bstride + m%seg
// Tile 128x64/WG, BK=32, 8 waves x (16x64 strip).
// ======================================================================
__global__ __launch_bounds__(256) void gemm_kernel(
    const unsigned short* __restrict__ A0, const unsigned short* __restrict__ A1,
    long lda,
    const float* __restrict__ Bw, long ldb,
    float* __restrict__ Cf, unsigned short* __restrict__ Cb, long ldc,
    const float* __restrict__ Radd, long ldr,
    int M, int N, int K,
    int a_seg, int a_bstride, int a_off,
    int c_seg, int c_bstride, int c_off) {
  __shared__ unsigned short As[2][128][34];   // [buf][m][k]
  __shared__ unsigned short Bs[2][64][34];    // [buf][n][k]

  const int tid  = threadIdx.x;
  const int wave = tid >> 5;
  const int lane = tid & 31;
  const int half = lane >> 4;
  const int ln   = lane & 15;
  const int m0 = blockIdx.y * 128;
  const int n0 = blockIdx.x * 64;

  v8f acc[4];
  #pragma unroll
  for (int t = 0; t < 4; ++t)
    #pragma unroll
    for (int e = 0; e < 8; ++e) acc[t][e] = 0.0f;

  const int ar_r = tid >> 1;
  const int ar_c = (tid & 1) * 16;
  const int am = m0 + ar_r;
  const long arow = (long)a_off + (long)(am / a_seg) * a_bstride + (am % a_seg);
  const int br_k = tid >> 3;
  const int br_c = (tid & 7) * 8;

  auto stage = [&](int k0, int buf) {
    if (!A1) {
      const uint4* sg = (const uint4*)(A0 + arow * lda + k0 + ar_c);
      uint4 q0 = sg[0], q1 = sg[1];
      unsigned int* dst = (unsigned int*)&As[buf][ar_r][ar_c];
      dst[0] = q0.x; dst[1] = q0.y; dst[2] = q0.z; dst[3] = q0.w;
      dst[4] = q1.x; dst[5] = q1.y; dst[6] = q1.z; dst[7] = q1.w;
    } else {
      const uint4* sg0 = (const uint4*)(A0 + arow * lda + k0 + ar_c);
      const uint4* sg1 = (const uint4*)(A1 + arow * lda + k0 + ar_c);
      unsigned short gs[16], us[16];
      ((uint4*)gs)[0] = sg0[0]; ((uint4*)gs)[1] = sg0[1];
      ((uint4*)us)[0] = sg1[0]; ((uint4*)us)[1] = sg1[1];
      unsigned short* dst = &As[buf][ar_r][ar_c];
      #pragma unroll
      for (int j = 0; j < 16; ++j)
        dst[j] = f2bf(gelu_tanh(bf2f(gs[j])) * bf2f(us[j]));
    }
    const float4* bs4 = (const float4*)(Bw + (long)(k0 + br_k) * ldb + n0 + br_c);
    float4 b0 = bs4[0], b1 = bs4[1];
    float bv[8] = {b0.x, b0.y, b0.z, b0.w, b1.x, b1.y, b1.z, b1.w};
    #pragma unroll
    for (int j = 0; j < 8; ++j) Bs[buf][br_c + j][br_k] = f2bf(bv[j]);
    // stream tiles 2 K-steps ahead into cache (global_prefetch_b8)
    if (k0 + 64 < K) {
      __builtin_prefetch(Bw + (long)(k0 + 64 + br_k) * ldb + n0 + br_c, 0, 1);
      __builtin_prefetch(A0 + arow * lda + k0 + 64 + ar_c, 0, 1);
    }
  };

  stage(0, 0);
  int p = 0;
  for (int k0 = 0; k0 < K; k0 += 32, p ^= 1) {
    __syncthreads();
    if (k0 + 32 < K) stage(k0 + 32, p ^ 1);

    // batched fragment loads -> back-to-back WMMAs
    FragA af;
    #pragma unroll
    for (int j = 0; j < 8; ++j) {
      int kk = ((j >> 2) << 4) + (half << 3) + ((j & 3) << 1);   // A layout
      af.u[j] = *(const unsigned int*)&As[p][wave * 16 + ln][kk];
    }
    FragA bfr[4];
    #pragma unroll
    for (int nt = 0; nt < 4; ++nt)
      #pragma unroll
      for (int j = 0; j < 8; ++j) {
        int kk = ((j >> 2) << 4) + (half << 3) + ((j & 3) << 1); // B layout
        bfr[nt].u[j] = *(const unsigned int*)&Bs[p][nt * 16 + ln][kk];
      }
    #pragma unroll
    for (int nt = 0; nt < 4; ++nt)
      acc[nt] = __builtin_amdgcn_wmma_f32_16x16x32_bf16(
          false, af.bf, false, bfr[nt].bf, (short)0, acc[nt], false, false);
  }

  #pragma unroll
  for (int nt = 0; nt < 4; ++nt) {
    #pragma unroll
    for (int r = 0; r < 8; ++r) {
      int m = m0 + wave * 16 + half * 8 + r;
      long crow = (long)c_off + (long)(m / c_seg) * c_bstride + (m % c_seg);
      int col = n0 + nt * 16 + ln;
      float v = acc[nt][r];
      if (Cf) {
        if (Radd) v += Radd[(long)m * ldr + col];
        Cf[crow * ldc + col] = v;
      } else {
        Cb[crow * ldc + col] = f2bf(v);
      }
    }
  }
}

// ======================================================================
// Flash attention, tanh softcap + mask, double-buffered K/V tiles.
//  Q: [B,S,NHEAD,HDIM] f32 (RoPE'd, *HD^-0.5); K,V: [B,S,NKVH,HDIM] f32
//  Out: [B*S, INNER] bf16
// grid = (S/128, NHEAD, B), block = 256
// ======================================================================
__global__ __launch_bounds__(256) void attn_kernel(
    const float* __restrict__ Q, const float* __restrict__ Kg,
    const float* __restrict__ Vg, const unsigned char* __restrict__ mask,
    unsigned short* __restrict__ Out) {
  __shared__ unsigned short Ks[2][32][132];     // [buf][s][hd]
  __shared__ unsigned short Vs[2][128][34];     // [buf][hd][s] transposed
  __shared__ unsigned short Ps[8][16 * 34];     // per-wave P staging [m][s]

  const int tid = threadIdx.x;
  const int wave = tid >> 5, lane = tid & 31, half = lane >> 4, ln = lane & 15;
  const int b = blockIdx.z, h = blockIdx.y, kh = h >> 1;   // GQA
  const int m0 = blockIdx.x * 128;

  FragA qf[4];
  {
    int trow = m0 + wave * 16 + ln;
    const float* qp = Q + (((long)b * S_TOT + trow) * NHEAD + h) * HDIM;
    #pragma unroll
    for (int f = 0; f < 4; ++f)
      #pragma unroll
      for (int j = 0; j < 8; ++j) {
        int hd = f * 32 + ((j >> 2) << 4) + (half << 3) + ((j & 3) << 1);
        float2 v = *(const float2*)(qp + hd);
        qf[f].h[2 * j]     = f2bf(v.x);
        qf[f].h[2 * j + 1] = f2bf(v.y);
      }
  }

  v8f o[8];
  #pragma unroll
  for (int nt = 0; nt < 8; ++nt)
    #pragma unroll
    for (int e = 0; e < 8; ++e) o[nt][e] = 0.0f;
  float mrow[8], lrow[8];
  #pragma unroll
  for (int r = 0; r < 8; ++r) { mrow[r] = -3.0e38f; lrow[r] = 0.0f; }

  const int lr = tid >> 3;
  const int lc = (tid & 7) * 16;

  auto stageKV = [&](int s0, int buf) {
    const float4* kp = (const float4*)(Kg + (((long)b * S_TOT + s0 + lr) * NKVH + kh) * HDIM + lc);
    const float4* vp = (const float4*)(Vg + (((long)b * S_TOT + s0 + lr) * NKVH + kh) * HDIM + lc);
    #pragma unroll
    for (int q4 = 0; q4 < 4; ++q4) {
      float4 kv = kp[q4], vv = vp[q4];
      int c = lc + q4 * 4;
      Ks[buf][lr][c]     = f2bf(kv.x); Ks[buf][lr][c + 1] = f2bf(kv.y);
      Ks[buf][lr][c + 2] = f2bf(kv.z); Ks[buf][lr][c + 3] = f2bf(kv.w);
      Vs[buf][c][lr]     = f2bf(vv.x); Vs[buf][c + 1][lr] = f2bf(vv.y);
      Vs[buf][c + 2][lr] = f2bf(vv.z); Vs[buf][c + 3][lr] = f2bf(vv.w);
    }
  };

  stageKV(0, 0);
  int p = 0;
  for (int s0 = 0; s0 < S_TOT; s0 += 32, p ^= 1) {
    __syncthreads();
    if (s0 + 32 < S_TOT) stageKV(s0 + 32, p ^ 1);

    // ---- logits: batched K-fragment loads, then 8 WMMAs ----
    FragA bk[4][2];
    #pragma unroll
    for (int f = 0; f < 4; ++f)
      #pragma unroll
      for (int j = 0; j < 8; ++j) {
        int hd = f * 32 + ((j >> 2) << 4) + (half << 3) + ((j & 3) << 1);
        bk[f][0].u[j] = *(const unsigned int*)&Ks[p][ln][hd];
        bk[f][1].u[j] = *(const unsigned int*)&Ks[p][16 + ln][hd];
      }
    v8f c0, c1;
    #pragma unroll
    for (int e = 0; e < 8; ++e) { c0[e] = 0.f; c1[e] = 0.f; }
    #pragma unroll
    for (int f = 0; f < 4; ++f) {
      c0 = __builtin_amdgcn_wmma_f32_16x16x32_bf16(false, qf[f].bf, false, bk[f][0].bf, (short)0, c0, false, false);
      c1 = __builtin_amdgcn_wmma_f32_16x16x32_bf16(false, qf[f].bf, false, bk[f][1].bf, (short)0, c1, false, false);
    }

    // ---- softcap + mask + online softmax ----
    const unsigned char* mbase =
        mask + ((long)b * S_TOT + (m0 + wave * 16 + half * 8)) * S_TOT + s0;
    float p0[8], p1[8], tmx[8];
    #pragma unroll
    for (int r = 0; r < 8; ++r) {
      float a0 = tanh_fast(c0[r] * 0.02f) * 50.0f;
      float a1 = tanh_fast(c1[r] * 0.02f) * 50.0f;
      const unsigned char* mp = mbase + (long)r * S_TOT;
      if (!mp[ln])      a0 = -1.0e30f;
      if (!mp[16 + ln]) a1 = -1.0e30f;
      c0[r] = a0; c1[r] = a1;
      tmx[r] = fmaxf(a0, a1);
    }
    #pragma unroll
    for (int r = 0; r < 8; ++r) {
      for (int off = 1; off < 16; off <<= 1)
        tmx[r] = fmaxf(tmx[r], __shfl_xor(tmx[r], off, 32));
      float mn = fmaxf(mrow[r], tmx[r]);
      float alpha = __expf(mrow[r] - mn);
      mrow[r] = mn;
      p0[r] = __expf(c0[r] - mn);
      p1[r] = __expf(c1[r] - mn);
      float ps = p0[r] + p1[r];
      for (int off = 1; off < 16; off <<= 1)
        ps += __shfl_xor(ps, off, 32);
      lrow[r] = lrow[r] * alpha + ps;
      #pragma unroll
      for (int nt = 0; nt < 8; ++nt) o[nt][r] *= alpha;
    }

    // ---- P (C-layout) -> LDS [m][s] -> A-fragment (same-wave, dscnt-ordered) ----
    unsigned short* pw = &Ps[wave][0];
    #pragma unroll
    for (int r = 0; r < 8; ++r) {
      int m = half * 8 + r;
      pw[m * 34 + ln]      = f2bf(p0[r]);
      pw[m * 34 + 16 + ln] = f2bf(p1[r]);
    }
    FragA ap;
    #pragma unroll
    for (int j = 0; j < 8; ++j) {
      int kk = ((j >> 2) << 4) + (half << 3) + ((j & 3) << 1);
      ap.u[j] = *(const unsigned int*)&pw[ln * 34 + kk];
    }
    // ---- o += P @ V: batched V-fragment loads, then 8 WMMAs ----
    FragA bv[8];
    #pragma unroll
    for (int nt = 0; nt < 8; ++nt)
      #pragma unroll
      for (int j = 0; j < 8; ++j) {
        int kk = ((j >> 2) << 4) + (half << 3) + ((j & 3) << 1);
        bv[nt].u[j] = *(const unsigned int*)&Vs[p][nt * 16 + ln][kk];
      }
    #pragma unroll
    for (int nt = 0; nt < 8; ++nt)
      o[nt] = __builtin_amdgcn_wmma_f32_16x16x32_bf16(false, ap.bf, false, bv[nt].bf, (short)0, o[nt], false, false);
  }

  #pragma unroll
  for (int r = 0; r < 8; ++r) {
    float inv = 1.0f / lrow[r];
    int trow = m0 + wave * 16 + half * 8 + r;
    unsigned short* op = Out + ((long)b * S_TOT + trow) * INNER + h * HDIM;
    #pragma unroll
    for (int nt = 0; nt < 8; ++nt)
      op[nt * 16 + ln] = f2bf(o[nt][r] * inv);
  }
}

// ======================================================================
// Host orchestration
// ======================================================================
extern "C" void kernel_launch(void* const* d_in, const int* in_sizes, int n_in,
                              void* d_out, int out_size, void* d_ws, size_t ws_size,
                              hipStream_t stream) {
  (void)in_sizes; (void)n_in; (void)out_size; (void)ws_size;

  const float* x_img        = (const float*)d_in[0];
  const float* x_txt        = (const float*)d_in[1];
  const unsigned char* mask = (const unsigned char*)d_in[2];
  const float* img_pre_attn = (const float*)d_in[3];
  const float* img_wq       = (const float*)d_in[4];
  const float* img_wk       = (const float*)d_in[5];
  const float* img_wv       = (const float*)d_in[6];
  const float* img_wo       = (const float*)d_in[7];
  const float* img_pre_ffw  = (const float*)d_in[8];
  const float* img_w_gate   = (const float*)d_in[9];
  const float* img_w_up     = (const float*)d_in[10];
  const float* img_w_down   = (const float*)d_in[11];
  const float* txt_pre_attn = (const float*)d_in[12];
  const float* txt_wq       = (const float*)d_in[13];
  const float* txt_wk       = (const float*)d_in[14];
  const float* txt_wv       = (const float*)d_in[15];
  const float* txt_wo       = (const float*)d_in[16];
  const float* txt_pre_ffw  = (const float*)d_in[17];
  const float* txt_w_gate   = (const float*)d_in[18];
  const float* txt_w_up     = (const float*)d_in[19];
  const float* txt_w_down   = (const float*)d_in[20];

  float* out_img = (float*)d_out;                                    // [2,1024,2048]
  float* out_txt = (float*)d_out + (size_t)B_SZ * LIMG * DIMG;       // [2,512,1024]

  char* ws = (char*)d_ws;
  size_t off = 0;
  auto alloc = [&](size_t bytes) -> void* {
    void* p = ws + off;
    off += (bytes + 255) & ~(size_t)255;
    return p;
  };
  unsigned short* xn_img = (unsigned short*)alloc((size_t)2048 * 2048 * 2);
  unsigned short* xn_txt = (unsigned short*)alloc((size_t)1024 * 1024 * 2);
  float* Qb  = (float*)alloc((size_t)3072 * 2048 * 4);   // [B*S, NH*HD]
  float* Kb  = (float*)alloc((size_t)3072 * 1024 * 4);   // [B*S, NKV*HD]
  float* Vb  = (float*)alloc((size_t)3072 * 1024 * 4);
  unsigned short* attn_o = (unsigned short*)alloc((size_t)3072 * 2048 * 2);
  float* proj_img = (float*)alloc((size_t)2048 * 2048 * 4);
  float* proj_txt = (float*)alloc((size_t)1024 * 1024 * 4);
  unsigned short* hn_img = (unsigned short*)alloc((size_t)2048 * 2048 * 2);
  unsigned short* hn_txt = (unsigned short*)alloc((size_t)1024 * 1024 * 2);
  unsigned short* gate_b = (unsigned short*)alloc((size_t)2048 * 8192 * 2);  // reused by txt
  unsigned short* up_b   = (unsigned short*)alloc((size_t)2048 * 8192 * 2);  // reused by txt

  // 1) pre-attention RMSNorms -> bf16
  rmsnorm_kernel<<<2048, 256, 0, stream>>>(x_img, nullptr, img_pre_attn, xn_img, 2048);
  rmsnorm_kernel<<<1024, 256, 0, stream>>>(x_txt, nullptr, txt_pre_attn, xn_txt, 1024);

  // 2) QKV projections (C remapped into joint [B,1536] layout)
  gemm_kernel<<<dim3(32, 16), 256, 0, stream>>>(xn_img, nullptr, 2048, img_wq, 2048,
      Qb, nullptr, 2048, nullptr, 0, 2048, 2048, 2048, 2048, 0, 0, 1024, 1536, 0);
  gemm_kernel<<<dim3(16, 16), 256, 0, stream>>>(xn_img, nullptr, 2048, img_wk, 1024,
      Kb, nullptr, 1024, nullptr, 0, 2048, 1024, 2048, 2048, 0, 0, 1024, 1536, 0);
  gemm_kernel<<<dim3(16, 16), 256, 0, stream>>>(xn_img, nullptr, 2048, img_wv, 1024,
      Vb, nullptr, 1024, nullptr, 0, 2048, 1024, 2048, 2048, 0, 0, 1024, 1536, 0);
  gemm_kernel<<<dim3(32, 8), 256, 0, stream>>>(xn_txt, nullptr, 1024, txt_wq, 2048,
      Qb, nullptr, 2048, nullptr, 0, 1024, 2048, 1024, 1024, 0, 0, 512, 1536, 1024);
  gemm_kernel<<<dim3(16, 8), 256, 0, stream>>>(xn_txt, nullptr, 1024, txt_wk, 1024,
      Kb, nullptr, 1024, nullptr, 0, 1024, 1024, 1024, 1024, 0, 0, 512, 1536, 1024);
  gemm_kernel<<<dim3(16, 8), 256, 0, stream>>>(xn_txt, nullptr, 1024, txt_wv, 1024,
      Vb, nullptr, 1024, nullptr, 0, 1024, 1024, 1024, 1024, 0, 0, 512, 1536, 1024);

  // 3) RoPE (+ q * HD^-0.5)
  rope_kernel<<<dim3(3072, 16), 64, 0, stream>>>(Qb, 16, 0.08838834764831845f);
  rope_kernel<<<dim3(3072, 8), 64, 0, stream>>>(Kb, 8, 1.0f);

  // 4) joint grouped-query flash attention
  attn_kernel<<<dim3(12, 16, 2), 256, 0, stream>>>(Qb, Kb, Vb, mask, attn_o);

  // 5) output projections (A remapped out of joint layout)
  gemm_kernel<<<dim3(32, 16), 256, 0, stream>>>(attn_o, nullptr, 2048, img_wo, 2048,
      proj_img, nullptr, 2048, nullptr, 0, 2048, 2048, 2048, 1024, 1536, 0, 2048, 0, 0);
  gemm_kernel<<<dim3(16, 8), 256, 0, stream>>>(attn_o, nullptr, 2048, txt_wo, 1024,
      proj_txt, nullptr, 1024, nullptr, 0, 1024, 1024, 2048, 512, 1536, 1024, 1024, 0, 0);

  // 6) add residual + pre-FFW RMSNorm
  rmsnorm_kernel<<<2048, 256, 0, stream>>>(proj_img, x_img, img_pre_ffw, hn_img, 2048);
  rmsnorm_kernel<<<1024, 256, 0, stream>>>(proj_txt, x_txt, txt_pre_ffw, hn_txt, 1024);

  // 7) image FFN: gate, up (bf16 out), fused gelu*up in down GEMM, + residual -> d_out
  gemm_kernel<<<dim3(128, 16), 256, 0, stream>>>(hn_img, nullptr, 2048, img_w_gate, 8192,
      nullptr, gate_b, 8192, nullptr, 0, 2048, 8192, 2048, 2048, 0, 0, 2048, 0, 0);
  gemm_kernel<<<dim3(128, 16), 256, 0, stream>>>(hn_img, nullptr, 2048, img_w_up, 8192,
      nullptr, up_b, 8192, nullptr, 0, 2048, 8192, 2048, 2048, 0, 0, 2048, 0, 0);
  gemm_kernel<<<dim3(32, 16), 256, 0, stream>>>(gate_b, up_b, 8192, img_w_down, 2048,
      out_img, nullptr, 2048, x_img, 2048, 2048, 2048, 8192, 2048, 0, 0, 2048, 0, 0);

  // 8) text FFN (reuses gate/up buffers)
  gemm_kernel<<<dim3(64, 8), 256, 0, stream>>>(hn_txt, nullptr, 1024, txt_w_gate, 4096,
      nullptr, gate_b, 4096, nullptr, 0, 1024, 4096, 1024, 1024, 0, 0, 1024, 0, 0);
  gemm_kernel<<<dim3(64, 8), 256, 0, stream>>>(hn_txt, nullptr, 1024, txt_w_up, 4096,
      nullptr, up_b, 4096, nullptr, 0, 1024, 4096, 1024, 1024, 0, 0, 1024, 0, 0);
  gemm_kernel<<<dim3(16, 8), 256, 0, stream>>>(gate_b, up_b, 4096, txt_w_down, 1024,
      out_txt, nullptr, 1024, x_txt, 1024, 1024, 1024, 4096, 1024, 0, 0, 1024, 0, 0);
}
